// NGPTAttention_84464826843532
// MI455X (gfx1250) — compile-verified, matrix operand order checked by reading, not run on hardware
//
#include <hip/hip_runtime.h>
#include <hip/hip_bf16.h>

// ---------------------------------------------------------------------------
// nGPT attention for MI455X (gfx1250):
//   - all GEMMs via v_wmma_f32_16x16x32_f16 (f32 accum)
//   - GEMM operands staged through LDS with global_load_async_to_lds_b128
//     (ASYNCcnt) and double buffering
//   - flash attention with online softmax, P transposed through LDS
// ---------------------------------------------------------------------------

#define S_LEN 2048
#define HID   2048
#define NH    16
#define NKVH  8
#define HD    128
#define QKV_N 4096            // H*D + KVH*D + KVH*D
#define SCALE 11.313708499f   // sqrt(128)
#define NEG_INF (-1e30f)

typedef __attribute__((ext_vector_type(16))) _Float16 v16h;
typedef __attribute__((ext_vector_type(8)))  _Float16 v8h;
typedef __attribute__((ext_vector_type(8)))  float    v8f;

union F16Frag { v16h v; v8h h[2]; };

// Load one 16-lane-wide f16 fragment (A-row or B-column, 32 K-values) honoring
// the CDNA5 16-bit A/B VGPR layout:
//   lanes 0-15 : K = [0..7] in VGPR0-3,  K = [16..23] in VGPR4-7
//   lanes 16-31: K = [8..15] in VGPR0-3, K = [24..31] in VGPR4-7
__device__ __forceinline__ v16h load_frag32(const _Float16* base, int sel) {
    F16Frag f;
    f.h[0] = *(const v8h*)(base + sel * 8);
    f.h[1] = *(const v8h*)(base + 16 + sel * 8);
    return f.v;
}

#define WMMA_F16(A, B, C) \
    __builtin_amdgcn_wmma_f32_16x16x32_f16(false, (A), false, (B), (short)0, (C), false, false)

// Async DMA: global (SADDR base + 32-bit byte offset, GVS mode) -> LDS.
// Tracked by ASYNCcnt; data bypasses VGPRs entirely.
__device__ __forceinline__ void async_b128(unsigned lds_byte, unsigned gl_byte,
                                           const void* sbase) {
    asm volatile("global_load_async_to_lds_b128 %0, %1, %2"
                 :: "v"(lds_byte), "v"(gl_byte), "s"(sbase)
                 : "memory");
}

// ---------------------------------------------------------------------------
// fp32 -> f16 conversion
// ---------------------------------------------------------------------------
__global__ __launch_bounds__(256) void cvt_f32_f16(const float* __restrict__ in,
                                                   _Float16* __restrict__ out, int n) {
    int i = blockIdx.x * 256 + threadIdx.x;
    if (i < n) out[i] = (_Float16)in[i];
}

// ---------------------------------------------------------------------------
// C[m,n] = sum_k A[m,k] * B[n,k]   (A: MxK row-major f16, B: NxK row-major f16,
// C: MxN fp32).  Block = 256 threads = 8 waves; block tile 64(M) x 128(N);
// each wave owns a 32x32 tile (4 wmma accumulators).
// K stepped by 32 with double-buffered async-to-LDS staging:
//   per stage each thread issues 3x global_load_async_to_lds_b128
//   (A tile 4KB + B tile 8KB), s_wait_asynccnt 3 overlaps the next stage's
//   DMA with this stage's ds_load_b128 + wmma.
// ---------------------------------------------------------------------------
__global__ __launch_bounds__(256) void gemm_f16_nt(const _Float16* __restrict__ A,
                                                   const _Float16* __restrict__ B,
                                                   float* __restrict__ C,
                                                   int M, int N, int K) {
    __shared__ _Float16 Alds[2][64 * 32];    // 2 x 4 KB
    __shared__ _Float16 Blds[2][128 * 32];   // 2 x 8 KB

    const int t    = threadIdx.x;
    const int lane = t & 31;
    const int wave = t >> 5;
    const int ln   = lane & 15;
    const int sel  = lane >> 4;
    const int wm   = wave & 1;
    const int wn   = wave >> 1;
    const int m0   = blockIdx.y * 64;
    const int n0   = blockIdx.x * 128;

    // staging assignment: A = 256 B128 ops (1/thread), B = 512 B128 ops (2/thread)
    const unsigned aRow = (unsigned)t >> 2, aSeg = (unsigned)t & 3;
    const unsigned bRow = (unsigned)t >> 1, bSeg = ((unsigned)t & 1) * 2u;
    const unsigned aGl  = ((unsigned)(m0 + (int)aRow) * (unsigned)K + aSeg * 8u) * 2u;
    const unsigned bGl  = ((unsigned)(n0 + (int)bRow) * (unsigned)K + bSeg * 8u) * 2u;
    const unsigned aLd  = aRow * 64u + aSeg * 16u;   // byte offset in 64x32 tile
    const unsigned bLd  = bRow * 64u + bSeg * 16u;   // byte offset in 128x32 tile
    const unsigned aBase[2] = { (unsigned)(size_t)(const void*)&Alds[0][0],
                                (unsigned)(size_t)(const void*)&Alds[1][0] };
    const unsigned bBase[2] = { (unsigned)(size_t)(const void*)&Blds[0][0],
                                (unsigned)(size_t)(const void*)&Blds[1][0] };

    v8f acc00 = {}, acc01 = {}, acc10 = {}, acc11 = {};

    // prime stage 0
    async_b128(aBase[0] + aLd,       aGl,       A);
    async_b128(bBase[0] + bLd,       bGl,       B);
    async_b128(bBase[0] + bLd + 16u, bGl + 16u, B);

    int buf = 0;
    for (int k0 = 0; k0 < K; k0 += 32) {
        if (k0 + 32 < K) {               // issue next stage, then drain previous
            const unsigned kB = (unsigned)(k0 + 32) * 2u;
            async_b128(aBase[buf ^ 1] + aLd,       aGl + kB,       A);
            async_b128(bBase[buf ^ 1] + bLd,       bGl + kB,       B);
            async_b128(bBase[buf ^ 1] + bLd + 16u, bGl + kB + 16u, B);
            asm volatile("s_wait_asynccnt 0x3" ::: "memory");
        } else {
            asm volatile("s_wait_asynccnt 0x0" ::: "memory");
        }
        __syncthreads();

        const _Float16* aL = &Alds[buf][(wm * 32 + ln) * 32];
        const _Float16* bL = &Blds[buf][(wn * 32 + ln) * 32];
        v16h a0 = load_frag32(aL, sel);
        v16h a1 = load_frag32(aL + 16 * 32, sel);
        v16h b0 = load_frag32(bL, sel);
        v16h b1 = load_frag32(bL + 16 * 32, sel);
        acc00 = WMMA_F16(a0, b0, acc00);
        acc01 = WMMA_F16(a0, b1, acc01);
        acc10 = WMMA_F16(a1, b0, acc10);
        acc11 = WMMA_F16(a1, b1, acc11);

        __syncthreads();   // all reads of buf done before it is DMA-overwritten
        buf ^= 1;
    }

    // C/D layout: VGPR r <-> row (r + 8*sel), lane&15 <-> column
    const int mw = m0 + wm * 32;
    const int nw = n0 + wn * 32;
    #pragma unroll
    for (int r = 0; r < 8; ++r) {
        int mr = r + 8 * sel;
        C[(size_t)(mw + mr) * N + nw + ln]           = acc00[r];
        C[(size_t)(mw + mr) * N + nw + 16 + ln]      = acc01[r];
        C[(size_t)(mw + 16 + mr) * N + nw + ln]      = acc10[r];
        C[(size_t)(mw + 16 + mr) * N + nw + 16 + ln] = acc11[r];
    }
}

// ---------------------------------------------------------------------------
// RoPE + L2 normalize + sqk^2 (q only) epilogue on the fp32 QKV GEMM output.
// grid = (S, 32): y<16 -> q head, 16<=y<24 -> k head, y>=24 -> v head (convert
// + transpose V to [kvh][d][s] so PV B-fragments are contiguous).
// ---------------------------------------------------------------------------
__global__ __launch_bounds__(128) void rope_norm_kernel(const float* __restrict__ QKV,
                                                        const float* __restrict__ cosp,
                                                        const float* __restrict__ sinp,
                                                        const float* __restrict__ sqkw,
                                                        _Float16* __restrict__ Qh,
                                                        _Float16* __restrict__ Kh,
                                                        _Float16* __restrict__ Vt) {
    __shared__ float sh[128];
    __shared__ float red[128];
    const int s = blockIdx.x;
    const int y = blockIdx.y;
    const int d = threadIdx.x;
    const float* row = QKV + (size_t)s * QKV_N;

    if (y >= 24) {                       // V: convert + transpose
        int hv = y - 24;
        float x = row[3072 + hv * HD + d];
        Vt[(size_t)(hv * HD + d) * S_LEN + s] = (_Float16)x;
        return;
    }
    const bool isQ = (y < 16);
    const int  h   = isQ ? y : (y - 16);
    float x = isQ ? row[h * HD + d] : row[2048 + h * HD + d];

    sh[d] = x;
    __syncthreads();
    float rot = (d < 64) ? -sh[d + 64] : sh[d - 64];
    float val = x * cosp[s * HD + d] + rot * sinp[s * HD + d];

    red[d] = val * val;
    __syncthreads();
    #pragma unroll
    for (int off = 64; off > 0; off >>= 1) {
        if (d < off) red[d] += red[d + off];
        __syncthreads();
    }
    val *= 1.0f / (sqrtf(red[0]) + 1e-8f);

    if (isQ) {
        float sq = sqkw[h * HD + d] * 50.0f;   // SQK_INIT_VALUE / SQK_INIT_SCALING
        val *= sq * sq;
        Qh[(size_t)s * (NH * HD) + h * HD + d] = (_Float16)val;
    } else {
        Kh[(size_t)s * (NKVH * HD) + h * HD + d] = (_Float16)val;
    }
}

// ---------------------------------------------------------------------------
// Flash attention: one wave per (head, 16-row q tile).  Keys stepped 32 at a
// time: QK^T via 8 wmma, causal mask + online softmax with half-wave shfl_xor
// row reductions (row stats align with the C-layout rows, so accumulator
// rescale is lane-local), P transposed C-layout -> A-layout through LDS,
// PV via 8 wmma into a 16x128 fp32 accumulator.
// ---------------------------------------------------------------------------
__global__ __launch_bounds__(32) void attn_kernel(const _Float16* __restrict__ Qh,
                                                  const _Float16* __restrict__ Kh,
                                                  const _Float16* __restrict__ Vt,
                                                  _Float16* __restrict__ Oh) {
    __shared__ _Float16 pLds[16 * 32];   // 1 KB P staging tile
    const int lane  = threadIdx.x;
    const int ln    = lane & 15;
    const int sel   = lane >> 4;
    const int h     = blockIdx.y;
    const int hk    = h >> 1;            // GQA: N_REP = 2
    const int qbase = blockIdx.x * 16;

    // Q fragments: 16 x 128 as four K=32 chunks
    v16h qa[4];
    {
        const _Float16* qrow = Qh + (size_t)(qbase + ln) * (NH * HD) + h * HD;
        #pragma unroll
        for (int kc = 0; kc < 4; ++kc) qa[kc] = load_frag32(qrow + kc * 32, sel);
    }

    v8f oacc[8];
    #pragma unroll
    for (int t = 0; t < 8; ++t) oacc[t] = (v8f){};
    float row_m[8], row_l[8];
    #pragma unroll
    for (int r = 0; r < 8; ++r) { row_m[r] = NEG_INF; row_l[r] = 0.0f; }

    const int kend = qbase + 16;                 // causal: kv <= q
    for (int kvb = 0; kvb < kend; kvb += 32) {
        v8f s0 = {}, s1 = {};
        const _Float16* krow0 = Kh + (size_t)(kvb + ln) * (NKVH * HD) + hk * HD;
        const _Float16* krow1 = krow0 + 16 * (NKVH * HD);
        #pragma unroll
        for (int kc = 0; kc < 4; ++kc) {
            v16h bk0 = load_frag32(krow0 + kc * 32, sel);
            v16h bk1 = load_frag32(krow1 + kc * 32, sel);
            s0 = WMMA_F16(qa[kc], bk0, s0);
            s1 = WMMA_F16(qa[kc], bk1, s1);
        }

        // online softmax, row-wise (row m = r + 8*sel, col = ln / ln+16)
        #pragma unroll
        for (int r = 0; r < 8; ++r) {
            const int qi  = qbase + r + 8 * sel;
            const int kv0 = kvb + ln;
            float x0 = (kv0      <= qi) ? s0[r] * SCALE : NEG_INF;
            float x1 = (kv0 + 16 <= qi) ? s1[r] * SCALE : NEG_INF;
            float mx = fmaxf(x0, x1);
            #pragma unroll
            for (int msk = 1; msk < 16; msk <<= 1) mx = fmaxf(mx, __shfl_xor(mx, msk, 32));
            const float nm = fmaxf(row_m[r], mx);
            const float p0 = __expf(x0 - nm);
            const float p1 = __expf(x1 - nm);
            float rs = p0 + p1;
            #pragma unroll
            for (int msk = 1; msk < 16; msk <<= 1) rs += __shfl_xor(rs, msk, 32);
            const float f = __expf(row_m[r] - nm);
            row_m[r] = nm;
            row_l[r] = row_l[r] * f + rs;
            #pragma unroll
            for (int t = 0; t < 8; ++t) oacc[t][r] *= f;
            const int m_ = r + 8 * sel;
            pLds[m_ * 32 + ln]      = (_Float16)p0;
            pLds[m_ * 32 + ln + 16] = (_Float16)p1;
        }

        // C-layout -> A-layout through LDS (same-wave LDS ops are in-order)
        asm volatile("" ::: "memory");
        v16h pa = load_frag32(pLds + ln * 32, sel);
        asm volatile("" ::: "memory");

        // PV: B column n = d-column of V (contiguous in transposed Vt)
        #pragma unroll
        for (int t = 0; t < 8; ++t) {
            const _Float16* vcol = Vt + (size_t)(hk * HD + t * 16 + ln) * S_LEN + kvb;
            v16h bv = load_frag32(vcol, sel);
            oacc[t] = WMMA_F16(pa, bv, oacc[t]);
        }
    }

    // epilogue: divide by row sums, emit f16 attention output [s][h][d]
    #pragma unroll
    for (int t = 0; t < 8; ++t) {
        #pragma unroll
        for (int r = 0; r < 8; ++r) {
            const int m_ = r + 8 * sel;
            float val = oacc[t][r] / row_l[r];
            Oh[(size_t)(qbase + m_) * (NH * HD) + h * HD + t * 16 + ln] = (_Float16)val;
        }
    }
}

// ---------------------------------------------------------------------------
// Host-side orchestration
// ---------------------------------------------------------------------------
extern "C" void kernel_launch(void* const* d_in, const int* in_sizes, int n_in,
                              void* d_out, int out_size, void* d_ws, size_t ws_size,
                              hipStream_t stream) {
    const float* hs   = (const float*)d_in[0];
    const float* cosp = (const float*)d_in[1];
    const float* sinp = (const float*)d_in[2];
    // d_in[3] attention_mask: causal mask applied analytically in attn_kernel
    const float* Wq   = (const float*)d_in[4];
    const float* Wk   = (const float*)d_in[5];
    const float* Wv   = (const float*)d_in[6];
    const float* Wo   = (const float*)d_in[7];
    const float* sqkw = (const float*)d_in[8];

    char* ws = (char*)d_ws;
    const size_t MB = 1024u * 1024u;
    _Float16* hsb   = (_Float16*)(ws + 0 * MB);    //  8 MiB: hs f16      (2048x2048)
    _Float16* wqkvb = (_Float16*)(ws + 8 * MB);    // 16 MiB: [Wq;Wk;Wv]  (4096x2048)
    _Float16* wob   = (_Float16*)(ws + 24 * MB);   //  8 MiB: Wo f16      (2048x2048)
    float*    qkvf  = (float*)   (ws + 32 * MB);   // 32 MiB: QKV fp32    (2048x4096)
    _Float16* qh    = (_Float16*)(ws + 64 * MB);   //  8 MiB: q f16       (2048x2048)
    _Float16* kh    = (_Float16*)(ws + 72 * MB);   //  4 MiB: k f16       (2048x1024)
    _Float16* vt    = (_Float16*)(ws + 76 * MB);   //  4 MiB: v^T f16     (1024x2048)
    _Float16* oh    = (_Float16*)(ws + 80 * MB);   //  8 MiB: attn out    (2048x2048)

    const int nHS = S_LEN * HID;        // 4194304
    const int nWq = NH * HD * HID;      // 4194304
    const int nWk = NKVH * HD * HID;    // 2097152

    // 1) fp32 -> f16 packs
    cvt_f32_f16<<<(nHS + 255) / 256, 256, 0, stream>>>(hs, hsb, nHS);
    cvt_f32_f16<<<(nWq + 255) / 256, 256, 0, stream>>>(Wq, wqkvb, nWq);
    cvt_f32_f16<<<(nWk + 255) / 256, 256, 0, stream>>>(Wk, wqkvb + nWq, nWk);
    cvt_f32_f16<<<(nWk + 255) / 256, 256, 0, stream>>>(Wv, wqkvb + nWq + nWk, nWk);
    cvt_f32_f16<<<(nWq + 255) / 256, 256, 0, stream>>>(Wo, wob, nWq);

    // 2) fused QKV projection: (2048x2048) x (4096x2048)^T -> 2048x4096 fp32
    gemm_f16_nt<<<dim3(QKV_N / 128, S_LEN / 64), 256, 0, stream>>>(
        hsb, wqkvb, qkvf, S_LEN, QKV_N, HID);

    // 3) RoPE + L2 norm + sqk^2, V transpose
    rope_norm_kernel<<<dim3(S_LEN, 32), 128, 0, stream>>>(
        qkvf, cosp, sinp, sqkw, qh, kh, vt);

    // 4) flash attention: one wave per (16-row q tile, head)
    attn_kernel<<<dim3(S_LEN / 16, NH), 32, 0, stream>>>(qh, kh, vt, oh);

    // 5) output projection: (2048x2048) x (2048x2048)^T -> fp32 d_out
    gemm_f16_nt<<<dim3(HID / 128, S_LEN / 64), 256, 0, stream>>>(
        oh, wob, (float*)d_out, S_LEN, HID, NH * HD);
}
